// SpatialLocalizationHead_59846074303072
// MI455X (gfx1250) — compile-verified, matrix operand order checked by reading, not run on hardware
//
#include <hip/hip_runtime.h>
#include <hip/hip_bf16.h>
#include <math.h>

// ---------------------------------------------------------------------------
// SpatialLocalizationHead for MI455X (gfx1250, wave32, WMMA).
//
// Key algebraic rewrite: single-query attention => K/V GEMMs collapse to
//   pass1 : per-token LN stats + 5 folded dot products   (reads 154 MB)
//   pass2 : attention-weighted token pooling + local sum (reads 154 MB, ~L2 hot)
// The remaining real GEMMs (ctx/out-proj/MLP, M=256) run on v_wmma_f32_16x16x32_bf16.
// Roofline: ~308 MB HBM traffic => ~13 us at 23.3 TB/s; WMMA work is trivial.
// ---------------------------------------------------------------------------

#define DEV __device__ __forceinline__

typedef __attribute__((ext_vector_type(16))) __bf16 v16bf;
typedef __attribute__((ext_vector_type(8)))  float  v8f;

// problem sizes (fixed by the reference)
constexpr int Bc   = 16;
constexpr int Tc   = 16;
constexpr int Sc   = 196;
constexpr int Dc   = 768;
constexpr int BTc  = Bc * Tc;            // 256
constexpr int NTOK = BTc * Sc;           // 50176
constexpr int NHc  = 4;
constexpr int Hc   = 256;
constexpr int H2c  = 128;
constexpr float SCALE = 0.07216878364870323f;   // 1/sqrt(192)

// workspace layout (float elements, all offsets multiples of 4)
constexpr size_t WFOLD  = 0;                                   // 5*768 folded weight vectors
constexpr size_t CONSTS = WFOLD + 5 * 768;                     // 16 scalars
constexpr size_t MARR   = CONSTS + 16;                         // per-token mean
constexpr size_t ISTD   = MARR + (size_t)NTOK;                 // per-token 1/std
constexpr size_t SCORES = ISTD + (size_t)NTOK;                 // [bt][h][s]
constexpr size_t OBJ    = SCORES + (size_t)BTc * NHc * Sc;     // [bt][s]
constexpr size_t ATTNU  = OBJ + (size_t)BTc * Sc;              // attn*istd  [bt][h][s]
constexpr size_t BETA   = ATTNU + (size_t)BTc * NHc * Sc;      // [bt][h]
constexpr size_t CXCY   = BETA + (size_t)BTc * NHc;            // [bt][2]
constexpr size_t IDX9   = CXCY + (size_t)BTc * 2;              // int, stride 12
constexpr size_t PFLAT  = IDX9 + (size_t)BTc * 12;             // pooled_flat [bt][h][c]
constexpr size_t CTXO   = PFLAT + (size_t)BTc * NHc * Dc;      // [bt][768]
constexpr size_t YBUF   = CTXO + (size_t)BTc * Dc;             // [bt][768]
constexpr size_t CATB   = YBUF + (size_t)BTc * Dc;             // [bt][1536] = pooled||local
constexpr size_t H1B    = CATB + (size_t)BTc * 2 * Dc;         // [bt][256]
constexpr size_t H2B    = H1B + (size_t)BTc * Hc;              // [bt][128]
constexpr size_t DLT    = H2B + (size_t)BTc * H2c;             // [bt][2]

DEV __bf16 tobf(float f) {
    union { float f; unsigned u; } in; in.f = f;
    unsigned r = in.u + 0x7FFFu + ((in.u >> 16) & 1u);          // RNE
    union { unsigned short s; __bf16 b; } out;
    out.s = (unsigned short)(r >> 16);
    return out.b;
}

DEV float gelu_exact(float x) { return 0.5f * x * (1.0f + erff(x * 0.70710678118654752f)); }

DEV float blk_reduce_sum(float v, float* red, int tid) {
    red[tid] = v; __syncthreads();
    for (int o = 128; o > 0; o >>= 1) { if (tid < o) red[tid] += red[tid + o]; __syncthreads(); }
    float r = red[0]; __syncthreads();
    return r;
}
DEV float blk_reduce_max(float v, float* red, int tid) {
    red[tid] = v; __syncthreads();
    for (int o = 128; o > 0; o >>= 1) { if (tid < o) red[tid] = fmaxf(red[tid], red[tid + o]); __syncthreads(); }
    float r = red[0]; __syncthreads();
    return r;
}

// ---------------------------------------------------------------------------
// Kernel 0: fold query/LN into per-token dot-product weight vectors.
//   wfold[h] = (1/sqrt(dh)) * ln_w ⊙ (Wk_hᵀ q_h),  wfold[4] = ln_w ⊙ obj_w
//   A_* = Σ wfold,  C_* = constant terms (ln_b·kq + q·bk, obj_b ...)
// ---------------------------------------------------------------------------
__global__ void __launch_bounds__(256) prep_kernel(
    const float* __restrict__ query, const float* __restrict__ ipw,
    const float* __restrict__ ipb, const float* __restrict__ ln_w,
    const float* __restrict__ ln_b, const float* __restrict__ obj_w,
    const float* __restrict__ obj_b, float* __restrict__ ws)
{
    __shared__ float qh[768];
    __shared__ float accA[5], accC[5];
    int tid = threadIdx.x;
    if (tid < 5) { accA[tid] = 0.f; accC[tid] = 0.f; }
    for (int j = tid; j < 768; j += 256) {
        float a = ipb[j];
        const float* wr = ipw + (size_t)j * 768;
        for (int c = 0; c < 768; ++c) a += wr[c] * query[c];
        qh[j] = a;
    }
    __syncthreads();
    for (int idx = tid; idx < 4 * 768; idx += 256) {
        int h = idx / 768, c = idx - h * 768;
        const float* col = ipw + (size_t)(768 + h * 192) * 768 + c;
        float kq = 0.f;
        for (int d = 0; d < 192; ++d) kq += qh[h * 192 + d] * col[(size_t)d * 768];
        float wfv = SCALE * ln_w[c] * kq;
        ws[WFOLD + (size_t)h * 768 + c] = wfv;
        atomicAdd(&accA[h], wfv);
        atomicAdd(&accC[h], SCALE * ln_b[c] * kq);
    }
    for (int c = tid; c < 768; c += 256) {
        float wo = ln_w[c] * obj_w[c];
        ws[WFOLD + 4 * 768 + c] = wo;
        atomicAdd(&accA[4], wo);
        atomicAdd(&accC[4], ln_b[c] * obj_w[c]);
    }
    __syncthreads();
    if (tid < 4) {
        float qb = 0.f;
        for (int d = 0; d < 192; ++d) qb += qh[tid * 192 + d] * ipb[768 + tid * 192 + d];
        ws[CONSTS + tid]     = accA[tid];
        ws[CONSTS + 4 + tid] = accC[tid] + SCALE * qb;
    }
    if (tid == 4) { ws[CONSTS + 8] = accA[4]; ws[CONSTS + 9] = accC[4] + obj_b[0]; }
}

// ---------------------------------------------------------------------------
// Kernel 1: streaming pass 1 — per-token LN stats + 5 folded dot products.
// One wave per token (24 f32 per lane via float4), wave shfl reductions.
// ---------------------------------------------------------------------------
__global__ void __launch_bounds__(256) pass1_kernel(
    const float* __restrict__ tokens, float* __restrict__ ws)
{
    __shared__ __align__(16) float wf[5 * 768];
    int tid = threadIdx.x;
    for (int i = tid; i < 5 * 768; i += 256) wf[i] = ws[WFOLD + i];
    __syncthreads();

    int wave = tid >> 5, lane = tid & 31;
    int tok  = blockIdx.x * 8 + wave;
    const float4* X   = (const float4*)(tokens + (size_t)tok * Dc);
    const float4* wf4 = (const float4*)wf;

    float s1 = 0.f, s2 = 0.f, d0 = 0.f, d1 = 0.f, d2 = 0.f, d3 = 0.f, d4 = 0.f;
#pragma unroll
    for (int j = 0; j < 6; ++j) {
        int idx = lane + 32 * j;
        float4 x = X[idx];
        s1 += x.x + x.y + x.z + x.w;
        s2 += x.x * x.x + x.y * x.y + x.z * x.z + x.w * x.w;
        float4 w0 = wf4[idx];
        float4 w1 = wf4[192 + idx];
        float4 w2 = wf4[384 + idx];
        float4 w3 = wf4[576 + idx];
        float4 w4 = wf4[768 + idx];
        d0 += x.x * w0.x + x.y * w0.y + x.z * w0.z + x.w * w0.w;
        d1 += x.x * w1.x + x.y * w1.y + x.z * w1.z + x.w * w1.w;
        d2 += x.x * w2.x + x.y * w2.y + x.z * w2.z + x.w * w2.w;
        d3 += x.x * w3.x + x.y * w3.y + x.z * w3.z + x.w * w3.w;
        d4 += x.x * w4.x + x.y * w4.y + x.z * w4.z + x.w * w4.w;
    }
    for (int o = 16; o > 0; o >>= 1) {
        s1 += __shfl_down(s1, o, 32);
        s2 += __shfl_down(s2, o, 32);
        d0 += __shfl_down(d0, o, 32);
        d1 += __shfl_down(d1, o, 32);
        d2 += __shfl_down(d2, o, 32);
        d3 += __shfl_down(d3, o, 32);
        d4 += __shfl_down(d4, o, 32);
    }
    if (lane == 0) {
        float m    = s1 * (1.f / 768.f);
        float var  = s2 * (1.f / 768.f) - m * m;
        float istd = rsqrtf(var + 1e-5f);
        ws[MARR + tok] = m;
        ws[ISTD + tok] = istd;
        int bt = tok / 196, s = tok - bt * 196;
        size_t sb = SCORES + (size_t)bt * 784 + s;
        ws[sb + 0 * 196] = istd * (d0 - m * ws[CONSTS + 0]) + ws[CONSTS + 4];
        ws[sb + 1 * 196] = istd * (d1 - m * ws[CONSTS + 1]) + ws[CONSTS + 5];
        ws[sb + 2 * 196] = istd * (d2 - m * ws[CONSTS + 2]) + ws[CONSTS + 6];
        ws[sb + 3 * 196] = istd * (d3 - m * ws[CONSTS + 3]) + ws[CONSTS + 7];
        ws[OBJ + (size_t)bt * 196 + s] = istd * (d4 - m * ws[CONSTS + 8]) + ws[CONSTS + 9];
    }
}

// ---------------------------------------------------------------------------
// Kernel 2a: attention softmax (196 keys) -> u = attn*istd, beta = Σ u*m.
// ---------------------------------------------------------------------------
__global__ void __launch_bounds__(256) attn_kernel(float* __restrict__ ws)
{
    __shared__ float red[256];
    __shared__ float sm[196], si[196];
    int tid = threadIdx.x, bt = blockIdx.x;
    if (tid < 196) {
        sm[tid] = ws[MARR + (size_t)bt * 196 + tid];
        si[tid] = ws[ISTD + (size_t)bt * 196 + tid];
    }
    __syncthreads();
    for (int h = 0; h < 4; ++h) {
        float sc = (tid < 196) ? ws[SCORES + (size_t)bt * 784 + h * 196 + tid] : -3.0e38f;
        float mx = blk_reduce_max(sc, red, tid);
        float e  = (tid < 196) ? expf(sc - mx) : 0.f;
        float ss = blk_reduce_sum(e, red, tid);
        float bm = 0.f;
        if (tid < 196) {
            float a = e / ss;
            float u = a * si[tid];
            ws[ATTNU + (size_t)bt * 784 + h * 196 + tid] = u;
            bm = u * sm[tid];
        }
        float beta = blk_reduce_sum(bm, red, tid);
        if (tid == 0) ws[BETA + (size_t)bt * 4 + h] = beta;
    }
}

// ---------------------------------------------------------------------------
// Kernel 2b: separable bilinear 14x14 -> 56x56, softmax*temp, centroid,
// 3x3 local-window indices.
// ---------------------------------------------------------------------------
__global__ void __launch_bounds__(256) ups_kernel(
    float* __restrict__ ws, const float* __restrict__ temperature)
{
    __shared__ float objs[196];
    __shared__ float up[3136];
    __shared__ float red[256];
    int tid = threadIdx.x, bt = blockIdx.x;
    if (tid < 196) objs[tid] = ws[OBJ + (size_t)bt * 196 + tid];
    __syncthreads();
    float temp = fmaxf(temperature[0], 1.0f);
    for (int i = tid; i < 3136; i += 256) {
        int u = i / 56, v = i - u * 56;
        float su = fmaxf((u + 0.5f) * 0.25f - 0.5f, 0.f);
        int i0u = (int)su; if (i0u > 13) i0u = 13;
        int i1u = i0u + 1; if (i1u > 13) i1u = 13;
        float w1u = su - (float)i0u, w0u = 1.f - w1u;
        float sv = fmaxf((v + 0.5f) * 0.25f - 0.5f, 0.f);
        int i0v = (int)sv; if (i0v > 13) i0v = 13;
        int i1v = i0v + 1; if (i1v > 13) i1v = 13;
        float w1v = sv - (float)i0v, w0v = 1.f - w1v;
        float val = w0u * (w0v * objs[i0u * 14 + i0v] + w1v * objs[i0u * 14 + i1v])
                  + w1u * (w0v * objs[i1u * 14 + i0v] + w1v * objs[i1u * 14 + i1v]);
        up[i] = val * temp;
    }
    __syncthreads();
    float lmax = -3.0e38f;
    for (int i = tid; i < 3136; i += 256) lmax = fmaxf(lmax, up[i]);
    float mx = blk_reduce_max(lmax, red, tid);
    float se = 0.f, sx = 0.f, sy = 0.f;
    for (int i = tid; i < 3136; i += 256) {
        int u = i / 56, v = i - u * 56;
        float e = expf(up[i] - mx);
        se += e;
        sx += e * ((float)v + 0.5f) * (1.f / 56.f);
        sy += e * ((float)u + 0.5f) * (1.f / 56.f);
    }
    se = blk_reduce_sum(se, red, tid);
    sx = blk_reduce_sum(sx, red, tid);
    sy = blk_reduce_sum(sy, red, tid);
    if (tid == 0) {
        float cx0 = sx / se, cy0 = sy / se;
        ws[CXCY + (size_t)bt * 2 + 0] = cx0;
        ws[CXCY + (size_t)bt * 2 + 1] = cy0;
        float cgx = fminf(fmaxf(cx0 * 14.f, 0.f), 13.f);
        float cgy = fminf(fmaxf(cy0 * 14.f, 0.f), 13.f);
        int cxg = (int)cgx, cyg = (int)cgy;
        int* iws = (int*)ws;
        for (int dy = 0; dy < 3; ++dy)
            for (int dx = 0; dx < 3; ++dx) {
                int gy = cyg + dy - 1; gy = gy < 0 ? 0 : (gy > 13 ? 13 : gy);
                int gx = cxg + dx - 1; gx = gx < 0 ? 0 : (gx > 13 ? 13 : gx);
                iws[IDX9 + (size_t)bt * 12 + dy * 3 + dx] = gy * 14 + gx;
            }
    }
}

// ---------------------------------------------------------------------------
// Kernel 3: streaming pass 2 — weighted pooling over raw tokens (folded LN)
// producing pooled_flat[bt][h][768] and local[bt][768] (-> CAT second half).
// ---------------------------------------------------------------------------
__global__ void __launch_bounds__(256) pass2_kernel(
    const float* __restrict__ tokens, const float* __restrict__ ln_w,
    const float* __restrict__ ln_b, float* __restrict__ ws)
{
    __shared__ float uu[784];
    __shared__ float wloc[196];
    __shared__ int   icnt[196];
    __shared__ float red[256];
    int tid = threadIdx.x, bt = blockIdx.x;
    for (int i = tid; i < 784; i += 256) uu[i] = ws[ATTNU + (size_t)bt * 784 + i];
    if (tid < 196) icnt[tid] = 0;
    __syncthreads();
    if (tid == 0) {
        const int* iws = (const int*)ws;
        for (int i = 0; i < 9; ++i) icnt[iws[IDX9 + (size_t)bt * 12 + i]]++;
    }
    __syncthreads();
    float gpart = 0.f;
    if (tid < 196) {
        float w = (float)icnt[tid] * ws[ISTD + (size_t)bt * 196 + tid] * (1.f / 9.f);
        wloc[tid] = w;
        gpart = w * ws[MARR + (size_t)bt * 196 + tid];
    }
    float gamma = blk_reduce_sum(gpart, red, tid);

    float a0[3] = {0, 0, 0}, a1[3] = {0, 0, 0}, a2[3] = {0, 0, 0};
    float a3[3] = {0, 0, 0}, al[3] = {0, 0, 0};
    const float* base = tokens + (size_t)bt * 196 * 768;
    for (int s = 0; s < 196; ++s) {
        const float* row = base + (size_t)s * 768;
        __builtin_prefetch(row + 2 * 768 + tid, 0, 1);   // global_prefetch_b8 into L2
        float u0 = uu[s], u1 = uu[196 + s], u2 = uu[392 + s], u3 = uu[588 + s];
        float wl = wloc[s];
#pragma unroll
        for (int j = 0; j < 3; ++j) {
            float x = row[tid + (j << 8)];
            a0[j] += u0 * x; a1[j] += u1 * x; a2[j] += u2 * x; a3[j] += u3 * x;
            al[j] += wl * x;
        }
    }
    float b0 = ws[BETA + (size_t)bt * 4 + 0];
    float b1 = ws[BETA + (size_t)bt * 4 + 1];
    float b2 = ws[BETA + (size_t)bt * 4 + 2];
    float b3 = ws[BETA + (size_t)bt * 4 + 3];
#pragma unroll
    for (int j = 0; j < 3; ++j) {
        int c = tid + (j << 8);
        float lw = ln_w[c], lb = ln_b[c];
        size_t pb = PFLAT + (size_t)bt * 3072 + c;
        ws[pb + 0 * 768] = lw * (a0[j] - b0) + lb;
        ws[pb + 1 * 768] = lw * (a1[j] - b1) + lb;
        ws[pb + 2 * 768] = lw * (a2[j] - b2) + lb;
        ws[pb + 3 * 768] = lw * (a3[j] - b3) + lb;
        ws[CATB + (size_t)bt * 1536 + 768 + c] = lw * (al[j] - gamma) + lb;
    }
}

// ---------------------------------------------------------------------------
// WMMA bf16 GEMM: C[m,n] = act(Σ_k A[m,k] * W[n,k] + bias[n]).
// One 16x16 tile per wave; K multiple of 32; M multiple of 16; N ragged (clamped
// W-row loads + guarded stores). Register packing per ISA 16-bit A layout
// (lane = m | (khalf<<4); VGPR v holds k = (v>=4)*16 + (v&3)*2 + khalf*8 .. +1);
// B assumed layout-symmetric with N in place of M. C: m = r + 8*khalf, n = lane&15.
// ---------------------------------------------------------------------------
__global__ void __launch_bounds__(128) gemm_wmma_bf16(
    const float* __restrict__ A, int lda, int strideA,
    const float* __restrict__ W, int ldw, int strideW,
    const float* __restrict__ bias, int strideB,
    float* __restrict__ C, int ldc, int strideC,
    int tilesM, int tilesN, int N, int K, int act)
{
    int wave = threadIdx.x >> 5;
    int tile = blockIdx.x * 4 + wave;
    if (tile >= tilesM * tilesN) return;
    int batch = blockIdx.y;
    int lane = threadIdx.x & 31;
    int hs   = lane >> 4;          // k-half select
    int lm   = lane & 15;
    int tm = tile / tilesN, tn = tile - tm * tilesN;
    int m0 = tm << 4, n0 = tn << 4;
    int arow = m0 + lm;
    int wrow = n0 + lm; if (wrow > N - 1) wrow = N - 1;
    const float* Ap = A + (size_t)batch * strideA + (size_t)arow * lda;
    const float* Wp = W + (size_t)batch * strideW + (size_t)wrow * ldw;
    v8f acc = {};
    for (int k0 = 0; k0 < K; k0 += 32) {
        v16bf av, wv;
#pragma unroll
        for (int v = 0; v < 8; ++v) {
            int kk = k0 + ((v & 4) << 2) + ((v & 3) << 1) + (hs << 3);
            float2 af = *(const float2*)(Ap + kk);
            float2 wf = *(const float2*)(Wp + kk);
            av[2 * v]     = tobf(af.x);
            av[2 * v + 1] = tobf(af.y);
            wv[2 * v]     = tobf(wf.x);
            wv[2 * v + 1] = tobf(wf.y);
        }
        acc = __builtin_amdgcn_wmma_f32_16x16x32_bf16(false, av, false, wv,
                                                      (short)0, acc, false, false);
    }
    int n = n0 + lm;
    if (n < N) {
        float bval = bias ? bias[(size_t)batch * strideB + n] : 0.f;
#pragma unroll
        for (int r = 0; r < 8; ++r) {
            int m = m0 + r + (hs << 3);
            float val = acc[r] + bval;
            if (act) val = gelu_exact(val);
            C[(size_t)batch * strideC + (size_t)m * ldc + n] = val;
        }
    }
}

// ---------------------------------------------------------------------------
// Kernel 6: rowwise LN of y -> CAT first half (pooled).
// ---------------------------------------------------------------------------
__global__ void __launch_bounds__(256) lnrow_kernel(
    const float* __restrict__ an_w, const float* __restrict__ an_b, float* __restrict__ ws)
{
    __shared__ float red[256];
    int tid = threadIdx.x, bt = blockIdx.x;
    const float* y = ws + YBUF + (size_t)bt * 768;
    float v0 = y[tid], v1 = y[tid + 256], v2 = y[tid + 512];
    float s1 = blk_reduce_sum(v0 + v1 + v2, red, tid);
    float s2 = blk_reduce_sum(v0 * v0 + v1 * v1 + v2 * v2, red, tid);
    float m = s1 * (1.f / 768.f);
    float istd = rsqrtf(s2 * (1.f / 768.f) - m * m + 1e-5f);
    float* cat = ws + CATB + (size_t)bt * 1536;
    cat[tid]       = (v0 - m) * istd * an_w[tid]       + an_b[tid];
    cat[tid + 256] = (v1 - m) * istd * an_w[tid + 256] + an_b[tid + 256];
    cat[tid + 512] = (v2 - m) * istd * an_w[tid + 512] + an_b[tid + 512];
}

// ---------------------------------------------------------------------------
// Kernel 10: center refinement, temporal convs (2->32->2, k=5, pad 2), boxes.
// ---------------------------------------------------------------------------
__global__ void __launch_bounds__(256) finale_kernel(
    const float* __restrict__ ws, const float* __restrict__ fixed_wh,
    const float* __restrict__ tc1_w, const float* __restrict__ tc1_b,
    const float* __restrict__ tc2_w, const float* __restrict__ tc2_b,
    float* __restrict__ out)
{
    __shared__ float sq[2][256];
    __shared__ float t1[16 * 32 * 16];
    __shared__ float td[16 * 2 * 16];
    int tid = threadIdx.x;
    {
        float cx0 = ws[CXCY + (size_t)tid * 2 + 0];
        float cy0 = ws[CXCY + (size_t)tid * 2 + 1];
        float d0  = ws[DLT + (size_t)tid * 2 + 0];
        float d1  = ws[DLT + (size_t)tid * 2 + 1];
        sq[0][tid] = fminf(fmaxf(cx0 + 0.3f * tanhf(d0), 0.f), 1.f);
        sq[1][tid] = fminf(fmaxf(cy0 + 0.3f * tanhf(d1), 0.f), 1.f);
    }
    __syncthreads();
    for (int i = tid; i < 16 * 32 * 16; i += 256) {
        int b = i >> 9, rem = i & 511;
        int o = rem >> 4, t = rem & 15;
        float acc = tc1_b[o];
#pragma unroll
        for (int k = 0; k < 5; ++k) {
            int tt = t + k - 2;
            if (tt >= 0 && tt < 16) {
                acc += sq[0][b * 16 + tt] * tc1_w[o * 10 + k];
                acc += sq[1][b * 16 + tt] * tc1_w[o * 10 + 5 + k];
            }
        }
        t1[i] = gelu_exact(acc);
    }
    __syncthreads();
    for (int i = tid; i < 512; i += 256) {
        int b = i >> 5, rem = i & 31;
        int oc = rem >> 4, t = rem & 15;
        float acc = tc2_b[oc];
        for (int ic = 0; ic < 32; ++ic) {
#pragma unroll
            for (int k = 0; k < 5; ++k) {
                int tt = t + k - 2;
                if (tt >= 0 && tt < 16)
                    acc += t1[(b << 9) + (ic << 4) + tt] * tc2_w[oc * 160 + ic * 5 + k];
            }
        }
        td[i] = acc;                 // [b][oc][t]
    }
    __syncthreads();
    {
        int b = tid >> 4, t = tid & 15;
        float cx = sq[0][tid], cy = sq[1][tid];
        cx = fminf(fmaxf(cx + 0.12f * tanhf(td[(b << 5) + t]), 0.f), 1.f);
        cy = fminf(fmaxf(cy + 0.12f * tanhf(td[(b << 5) + 16 + t]), 0.f), 1.f);
        const float mins = 1.f / 14.f;
        float w = fminf(fmaxf(fixed_wh[0], mins), 1.f);
        float h = fminf(fmaxf(fixed_wh[1], mins), 1.f);
        float x1 = fminf(fmaxf(cx - 0.5f * w, 0.f), 1.f);
        float y1 = fminf(fmaxf(cy - 0.5f * h, 0.f), 1.f);
        float x2 = fminf(fmaxf(cx + 0.5f * w, 0.f), 1.f);
        float y2 = fminf(fmaxf(cy + 0.5f * h, 0.f), 1.f);
        x2 = fminf(fmaxf(fmaxf(x2, x1 + mins), 0.f), 1.f);
        y2 = fminf(fmaxf(fmaxf(y2, y1 + mins), 0.f), 1.f);
        out[tid * 4 + 0] = x1; out[tid * 4 + 1] = y1;
        out[tid * 4 + 2] = x2; out[tid * 4 + 3] = y2;
    }
}

extern "C" void kernel_launch(void* const* d_in, const int* in_sizes, int n_in,
                              void* d_out, int out_size, void* d_ws, size_t ws_size,
                              hipStream_t stream)
{
    (void)in_sizes; (void)n_in; (void)out_size; (void)ws_size;
    const float* tokens      = (const float*)d_in[0];
    const float* fixed_wh    = (const float*)d_in[2];
    const float* ln_w        = (const float*)d_in[3];
    const float* ln_b        = (const float*)d_in[4];
    const float* query       = (const float*)d_in[5];
    const float* ipw         = (const float*)d_in[6];
    const float* ipb         = (const float*)d_in[7];
    const float* out_w       = (const float*)d_in[8];
    const float* out_b       = (const float*)d_in[9];
    const float* an_w        = (const float*)d_in[10];
    const float* an_b        = (const float*)d_in[11];
    const float* obj_w       = (const float*)d_in[12];
    const float* obj_b       = (const float*)d_in[13];
    const float* temperature = (const float*)d_in[14];
    const float* r1_w        = (const float*)d_in[15];
    const float* r1_b        = (const float*)d_in[16];
    const float* r2_w        = (const float*)d_in[17];
    const float* r2_b        = (const float*)d_in[18];
    const float* r3_w        = (const float*)d_in[19];
    const float* r3_b        = (const float*)d_in[20];
    const float* tc1_w       = (const float*)d_in[21];
    const float* tc1_b       = (const float*)d_in[22];
    const float* tc2_w       = (const float*)d_in[23];
    const float* tc2_b       = (const float*)d_in[24];
    float* ws  = (float*)d_ws;
    float* out = (float*)d_out;

    prep_kernel<<<1, 256, 0, stream>>>(query, ipw, ipb, ln_w, ln_b, obj_w, obj_b, ws);
    pass1_kernel<<<NTOK / 8, 256, 0, stream>>>(tokens, ws);
    attn_kernel<<<BTc, 256, 0, stream>>>(ws);
    ups_kernel<<<BTc, 256, 0, stream>>>(ws, temperature);
    pass2_kernel<<<BTc, 256, 0, stream>>>(tokens, ln_w, ln_b, ws);

    // GEMM 1: ctx_h = pooled_flat_h @ Wv_h^T + bv_h   (batched over 4 heads)
    {
        dim3 grid((16 * 12 + 3) / 4, 4);
        gemm_wmma_bf16<<<grid, 128, 0, stream>>>(
            ws + PFLAT, 3072, 768,
            ipw + 2 * 768 * 768, 768, 192 * 768,
            ipb + 2 * 768, 192,
            ws + CTXO, 768, 192,
            16, 12, 192, 768, 0);
    }
    // GEMM 2: y = ctx @ out_w^T + out_b
    {
        dim3 grid((16 * 48 + 3) / 4, 1);
        gemm_wmma_bf16<<<grid, 128, 0, stream>>>(
            ws + CTXO, 768, 0, out_w, 768, 0, out_b, 0,
            ws + YBUF, 768, 0, 16, 48, 768, 768, 0);
    }
    lnrow_kernel<<<BTc, 256, 0, stream>>>(an_w, an_b, ws);
    // GEMM 3: h1 = gelu(cat @ r1_w^T + r1_b)
    {
        dim3 grid((16 * 16 + 3) / 4, 1);
        gemm_wmma_bf16<<<grid, 128, 0, stream>>>(
            ws + CATB, 1536, 0, r1_w, 1536, 0, r1_b, 0,
            ws + H1B, 256, 0, 16, 16, 256, 1536, 1);
    }
    // GEMM 4: h2 = gelu(h1 @ r2_w^T + r2_b)
    {
        dim3 grid((16 * 8 + 3) / 4, 1);
        gemm_wmma_bf16<<<grid, 128, 0, stream>>>(
            ws + H1B, 256, 0, r2_w, 256, 0, r2_b, 0,
            ws + H2B, 128, 0, 16, 8, 128, 256, 1);
    }
    // GEMM 5: delta = h2 @ r3_w^T + r3_b  (N=2, ragged tile)
    {
        dim3 grid((16 * 1 + 3) / 4, 1);
        gemm_wmma_bf16<<<grid, 128, 0, stream>>>(
            ws + H2B, 128, 0, r3_w, 128, 0, r3_b, 0,
            ws + DLT, 2, 0, 16, 1, 2, 128, 0);
    }
    finale_kernel<<<1, 256, 0, stream>>>(ws, fixed_wh, tc1_w, tc1_b, tc2_w, tc2_b, out);
}
